// FusedAttention_54090818126325
// MI455X (gfx1250) — matrix operand order repaired and checked by reading, not compiled, not run on hardware
//
#include <hip/hip_runtime.h>
#include <hip/hip_bf16.h>

// ---------------------------------------------------------------------------
// Types for CDNA5 WMMA (wave32): v_wmma_f32_16x16x32_bf16
// ---------------------------------------------------------------------------
typedef __bf16 bf16;
typedef __attribute__((ext_vector_type(8)))  __bf16 v8bf;
typedef __attribute__((ext_vector_type(16))) __bf16 v16bf;
typedef __attribute__((ext_vector_type(8)))  float  v8f;

__device__ __forceinline__ v16bf cat16(v8bf lo, v8bf hi) {
  return __builtin_shufflevector(lo, hi, 0,1,2,3,4,5,6,7,8,9,10,11,12,13,14,15);
}
__device__ __forceinline__ v8f wmma_bf16(v16bf a, v16bf b, v8f c) {
  // 8 args: (neg_a, A, neg_b, B, c_mod, C, reuse_a, reuse_b)
  return __builtin_amdgcn_wmma_f32_16x16x32_bf16(false, a, false, b, (short)0, c,
                                                 false, false);
}
__device__ __forceinline__ v8f v8f_zero() {
  v8f z;
#pragma unroll
  for (int i = 0; i < 8; ++i) z[i] = 0.0f;
  return z;
}
__device__ __forceinline__ void store_out(float* p, float v) { *p = v; }
__device__ __forceinline__ void store_out(bf16*  p, float v) { *p = (bf16)v; }

// Async global -> LDS copy, 16 bytes per lane (GLOBAL_LOAD_ASYNC_TO_LDS_B128,
// tracked by ASYNCcnt).  lds: per-lane LDS byte address (u32), g: per-lane
// 64-bit global address.
__device__ __forceinline__ void async_copy_b128(unsigned int lds, const void* g) {
  asm volatile("global_load_async_to_lds_b128 %0, %1, off"
               :: "v"(lds), "v"(g) : "memory");
}
__device__ __forceinline__ void wait_async0() {
  asm volatile("s_wait_asynccnt 0x0" ::: "memory");
}

// ---------------------------------------------------------------------------
// Kernel 0: fp32 -> bf16 conversion (grid-stride)
// ---------------------------------------------------------------------------
__global__ void cvt_f32_to_bf16(const float* __restrict__ src,
                                bf16* __restrict__ dst, long long n) {
  long long i = (long long)blockIdx.x * blockDim.x + threadIdx.x;
  long long stride = (long long)gridDim.x * blockDim.x;
  for (; i < n; i += stride) dst[i] = (bf16)src[i];
}

// ---------------------------------------------------------------------------
// Kernel 1/3: bf16 GEMM, C(MxN) = A(MxK) @ B(KxN), fp32 accumulate via WMMA.
// Block tile 128x128, BK=32, 256 threads = 8 waves arranged 4(M) x 2(N);
// each wave computes a 32x64 tile = 2x4 WMMA accumulators.
// Double-buffered LDS staged with async global->LDS copies.
// ---------------------------------------------------------------------------
template <typename OutT>
__global__ __launch_bounds__(256)
void gemm_bf16_wmma(const bf16* __restrict__ A, const bf16* __restrict__ B,
                    OutT* __restrict__ C, int M, int N, int K) {
  __shared__ bf16 As[2][128 * 32];   // [buf][row][k]   2 x 8 KB
  __shared__ bf16 Bs[2][32 * 128];   // [buf][k][col]   2 x 8 KB

  const int tid  = threadIdx.x;
  const int lane = tid & 31;
  const int wave = tid >> 5;
  const int wm   = wave >> 1;           // 0..3
  const int wn   = wave & 1;            // 0..1
  const int m0   = wm * 32;
  const int n0   = wn * 64;
  const int lh   = lane >> 4;           // 0/1  (half-wave)
  const int ll   = lane & 15;
  const int blockM = blockIdx.y * 128;
  const int blockN = blockIdx.x * 128;

  v8f acc[2][4];
#pragma unroll
  for (int mi = 0; mi < 2; ++mi)
#pragma unroll
    for (int ni = 0; ni < 4; ++ni) acc[mi][ni] = v8f_zero();

  // stage one 128x32 A-tile and 32x128 B-tile into LDS buffer `buf`
  auto stage = [&](int buf, int k0) {
#pragma unroll
    for (int i = 0; i < 2; ++i) {
      int c  = tid + i * 256;        // chunk id 0..511
      int r  = c >> 2;               // 0..127
      int c8 = (c & 3) * 8;          // 0,8,16,24
      int gr = blockM + r; if (gr >= M) gr = M - 1;
      async_copy_b128((unsigned int)(unsigned long long)&As[buf][r * 32 + c8],
                      A + (long long)gr * K + k0 + c8);
    }
#pragma unroll
    for (int i = 0; i < 2; ++i) {
      int c  = tid + i * 256;
      int r  = c >> 4;               // 0..31
      int c8 = (c & 15) * 8;         // 0..120
      async_copy_b128((unsigned int)(unsigned long long)&Bs[buf][r * 128 + c8],
                      B + (long long)(k0 + r) * N + blockN + c8);
    }
  };

  stage(0, 0);
  wait_async0();
  __syncthreads();

  int cur = 0;
  for (int k0 = 0; k0 < K; k0 += 32) {
    // kick off next tile into the other buffer while computing this one
    if (k0 + 32 < K) stage(cur ^ 1, k0 + 32);

    // ---- A fragments: lane<16 holds K {0..7,16..23}, lane>=16 {8..15,24..31}
    v16bf af[2];
#pragma unroll
    for (int mi = 0; mi < 2; ++mi) {
      const bf16* base = &As[cur][(m0 + mi * 16 + ll) * 32 + 8 * lh];
      v8bf lo = *(const v8bf*)base;          // K = 8*lh .. +7
      v8bf hi = *(const v8bf*)(base + 16);   // K = 16+8*lh .. +7
      af[mi] = cat16(lo, hi);
    }
    // ---- B fragments: lane L holds column N=L%16, K in table order ----
    v16bf bfr[4];
#pragma unroll
    for (int ni = 0; ni < 4; ++ni) {
#pragma unroll
      for (int s = 0; s < 16; ++s) {
        int kr = (s < 8) ? (8 * lh + s) : (16 + 8 * lh + (s - 8));
        bfr[ni][s] = Bs[cur][kr * 128 + n0 + ni * 16 + ll];
      }
    }
#pragma unroll
    for (int mi = 0; mi < 2; ++mi)
#pragma unroll
      for (int ni = 0; ni < 4; ++ni)
        acc[mi][ni] = wmma_bf16(af[mi], bfr[ni], acc[mi][ni]);

    wait_async0();       // our async stores into buf^1 are in LDS
    __syncthreads();     // everyone's are, and buf `cur` is fully consumed
    cur ^= 1;
  }

  // ---- epilogue: C/D layout -> element (m = v + 8*lh, n = ll) ----
#pragma unroll
  for (int mi = 0; mi < 2; ++mi)
#pragma unroll
    for (int ni = 0; ni < 4; ++ni) {
      int n = blockN + n0 + ni * 16 + ll;
#pragma unroll
      for (int v = 0; v < 8; ++v) {
        int m = blockM + m0 + mi * 16 + v + 8 * lh;
        if (m < M) store_out(C + (long long)m * N + n, acc[mi][ni][v]);
      }
    }
}

// ---------------------------------------------------------------------------
// Kernel 2: flash attention (bf16 operands, fp32 softmax/accumulate).
// Hybrid mask: causal + row 0 attends to everything.
// Block = 128 threads = 4 waves; each wave owns 16 query rows (64 rows/block).
// Key tiles of 32; S = Q K^T (4 WMMAs), O += P V (4 WMMAs).
// ---------------------------------------------------------------------------
__global__ __launch_bounds__(128)
void flash_attn_wmma(const bf16* __restrict__ qkv, bf16* __restrict__ y,
                     int Bn, int T) {
  const int C = 1024, D = 64, C3 = 3072, H = 16;
  const int nblk = (T + 63) / 64;

  const int blk = blockIdx.x % nblk;
  const int bh  = blockIdx.x / nblk;
  const int b   = bh / H;
  const int h   = bh % H;
  const int t0  = blk * 64;

  const int tid  = threadIdx.x;
  const int lane = tid & 31;
  const int wave = tid >> 5;
  const int lh   = lane >> 4;
  const int ll   = lane & 15;

  __shared__ bf16 Kt[64 * 32];        // transposed K tile [d][s], 4 KB
  __shared__ bf16 Pl[4 * 16 * 32];    // per-wave P staging, 4 KB

  // ---- load Q fragments (A-layout) once; reused for every key tile ----
  int tq = t0 + wave * 16 + ll; if (tq >= T) tq = T - 1;
  const bf16* qrow = qkv + (long long)(b * T + tq) * C3 + h * D;
  v16bf aQ[2];
#pragma unroll
  for (int kk = 0; kk < 2; ++kk) {
    const bf16* p = qrow + kk * 32 + 8 * lh;
    aQ[kk] = cat16(*(const v8bf*)p, *(const v8bf*)(p + 16));
  }

  v8f O[4];
#pragma unroll
  for (int dt = 0; dt < 4; ++dt) O[dt] = v8f_zero();
  float mrow[8], lrow[8];
#pragma unroll
  for (int v = 0; v < 8; ++v) { mrow[v] = -__builtin_inff(); lrow[v] = 0.0f; }

  const float scale = 0.125f;  // 1/sqrt(64)
  int smax = (t0 == 0) ? (T - 1) : ((t0 + 63 < T - 1) ? t0 + 63 : T - 1);
  const int ntiles = smax / 32 + 1;

  for (int it = 0; it < ntiles; ++it) {
    const int s0 = it * 32;

    // ---- stage K tile transposed: Kt[d][s], 128 threads ----
    {
      int s  = tid & 31;
      int dc = (tid >> 5) * 16;                 // 0,16,32,48
      int sk = s0 + s; if (sk >= T) sk = T - 1;
      const bf16* krow = qkv + (long long)(b * T + sk) * C3 + C + h * D + dc;
      v8bf k0v = *(const v8bf*)krow;
      v8bf k1v = *(const v8bf*)(krow + 8);
#pragma unroll
      for (int j = 0; j < 8; ++j) {
        Kt[(dc + j)     * 32 + s] = k0v[j];
        Kt[(dc + 8 + j) * 32 + s] = k1v[j];
      }
      // prefetch next tile's K row (global_prefetch_b8)
      if (s0 + 32 <= smax) {
        int sn = s0 + 32 + s; if (sn >= T) sn = T - 1;
        __builtin_prefetch(qkv + (long long)(b * T + sn) * C3 + C + h * D, 0, 1);
      }
    }
    __syncthreads();

    // ---- S = Q K^T : 16 rows x 32 keys in two accumulators ----
    v8f S0 = v8f_zero(), S1 = v8f_zero();
#pragma unroll
    for (int kk = 0; kk < 2; ++kk) {
      v16bf b0, b1;     // lane = key column ll, elements = d in table order
#pragma unroll
      for (int s = 0; s < 16; ++s) {
        int kr = (s < 8) ? (8 * lh + s) : (16 + 8 * lh + (s - 8));
        b0[s] = Kt[(kk * 32 + kr) * 32 + ll];
        b1[s] = Kt[(kk * 32 + kr) * 32 + 16 + ll];
      }
      S0 = wmma_bf16(aQ[kk], b0, S0);
      S1 = wmma_bf16(aQ[kk], b1, S1);
    }

    // ---- online softmax in C/D layout: row m = v + 8*lh, col = ll ----
    float p0[8], p1[8];
#pragma unroll
    for (int v = 0; v < 8; ++v) {
      int tm = t0 + wave * 16 + v + 8 * lh;
      int sA = s0 + ll, sB = s0 + 16 + ll;
      float x0 = S0[v] * scale, x1 = S1[v] * scale;
      bool ok0 = (sA < T) && ((sA <= tm) || (tm == 0));
      bool ok1 = (sB < T) && ((sB <= tm) || (tm == 0));
      x0 = ok0 ? x0 : -1.0e30f;
      x1 = ok1 ? x1 : -1.0e30f;
      float mx = fmaxf(x0, x1);
#pragma unroll
      for (int off = 8; off >= 1; off >>= 1)
        mx = fmaxf(mx, __shfl_xor(mx, off, 32));       // within 16-lane group
      float mnew = fmaxf(mrow[v], mx);
      float corr = __expf(mrow[v] - mnew);
      float e0 = __expf(x0 - mnew);
      float e1 = __expf(x1 - mnew);
      float rs = e0 + e1;
#pragma unroll
      for (int off = 8; off >= 1; off >>= 1)
        rs += __shfl_xor(rs, off, 32);
      lrow[v] = lrow[v] * corr + rs;
      mrow[v] = mnew;
#pragma unroll
      for (int dt = 0; dt < 4; ++dt) O[dt][v] *= corr;
      p0[v] = e0; p1[v] = e1;
    }

    // ---- re-layout P (C/D) -> A-matrix via per-wave LDS ----
    bf16* Pw = &Pl[wave * 16 * 32];
#pragma unroll
    for (int v = 0; v < 8; ++v) {
      int m = v + 8 * lh;
      Pw[m * 32 + ll]      = (bf16)p0[v];
      Pw[m * 32 + 16 + ll] = (bf16)p1[v];
    }
    asm volatile("" ::: "memory");   // keep DS store->load order (in-order per wave)
    v16bf aP;
    {
      const bf16* pr = &Pw[ll * 32 + 8 * lh];
      aP = cat16(*(const v8bf*)pr, *(const v8bf*)(pr + 16));
    }

    // ---- O += P V : V read directly in B-layout (lane = d column, elements = keys) ----
    {
#pragma unroll
      for (int dt = 0; dt < 4; ++dt) {
        v16bf bV;
#pragma unroll
        for (int s = 0; s < 16; ++s) {
          int kr = (s < 8) ? (8 * lh + s) : (16 + 8 * lh + (s - 8));
          int sk = s0 + kr; if (sk >= T) sk = T - 1;
          bV[s] = qkv[(long long)(b * T + sk) * C3 + 2 * C + h * D + dt * 16 + ll];
        }
        O[dt] = wmma_bf16(aP, bV, O[dt]);
      }
    }
    __syncthreads();   // Kt safe to overwrite next iteration
  }

  // ---- normalize and store y (bf16) ----
#pragma unroll
  for (int v = 0; v < 8; ++v) {
    int tm = t0 + wave * 16 + v + 8 * lh;
    if (tm < T) {
      float inv = 1.0f / lrow[v];
#pragma unroll
      for (int dt = 0; dt < 4; ++dt)
        y[(long long)(b * T + tm) * C + h * D + dt * 16 + ll] =
            (bf16)(O[dt][v] * inv);
    }
  }
}

// ---------------------------------------------------------------------------
// Host-side launcher
// ---------------------------------------------------------------------------
extern "C" void kernel_launch(void* const* d_in, const int* in_sizes, int n_in,
                              void* d_out, int out_size, void* d_ws, size_t ws_size,
                              hipStream_t stream) {
  (void)in_sizes; (void)n_in; (void)out_size; (void)ws_size;
  const float* x  = (const float*)d_in[0];   // (B,T,C)
  const float* Wa = (const float*)d_in[1];   // (C,3C)
  const float* Wp = (const float*)d_in[2];   // (C,C)
  float* out = (float*)d_out;                // (B,T,C) fp32

  const int B = 4, T = 2049, C = 1024, H = 16;
  const int M  = B * T;        // 8196
  const int C3 = 3 * C;        // 3072

  // workspace carve-up (all sizes 256B-multiples)
  char* ws = (char*)d_ws;
  bf16* xb   = (bf16*)ws;                 ws += (long long)M * C  * 2;  // 16.8 MB
  bf16* qkvb = (bf16*)ws;                 ws += (long long)M * C3 * 2;  // 50.4 MB
  bf16* yb   = (bf16*)ws;                 ws += (long long)M * C  * 2;  // 16.8 MB
  bf16* Wab  = (bf16*)ws;                 ws += (long long)C * C3 * 2;  //  6.3 MB
  bf16* Wpb  = (bf16*)ws;                                               //  2.1 MB

  // 0) fp32 -> bf16 conversions
  cvt_f32_to_bf16<<<4096, 256, 0, stream>>>(x,  xb,  (long long)M * C);
  cvt_f32_to_bf16<<<4096, 256, 0, stream>>>(Wa, Wab, (long long)C * C3);
  cvt_f32_to_bf16<<<2048, 256, 0, stream>>>(Wp, Wpb, (long long)C * C);

  // 1) qkv = x @ W_attn   (M x 3C), bf16 out
  {
    dim3 grid(C3 / 128, (M + 127) / 128);
    gemm_bf16_wmma<bf16><<<grid, 256, 0, stream>>>(xb, Wab, qkvb, M, C3, C);
  }

  // 2) flash attention -> yb (bf16)
  {
    int nblk = (T + 63) / 64;             // 33
    dim3 grid(B * H * nblk);              // 2112 blocks
    flash_attn_wmma<<<grid, 128, 0, stream>>>(qkvb, yb, B, T);
  }

  // 3) out = y @ W_proj   (M x C), fp32 out
  {
    dim3 grid(C / 128, (M + 127) / 128);
    gemm_bf16_wmma<float><<<grid, 256, 0, stream>>>(yb, Wpb, out, M, C, C);
  }
}
